// GCN_CN_v4_AE_bn32_twoStage_68204080660834
// MI455X (gfx1250) — matrix operand order repaired and checked by reading, not compiled, not run on hardware
//
#include <hip/hip_runtime.h>
#include <hip/hip_bf16.h>

typedef __attribute__((ext_vector_type(16))) _Float16 v16h;
typedef __attribute__((ext_vector_type(8)))  _Float16 v8h;
typedef __attribute__((ext_vector_type(8)))  float    v8f;

#define N_NODES 8192
#define N_EDGES 262144

static __device__ __forceinline__ void atomAddF32(float* p, float v) {
    unsafeAtomicAdd(p, v);
}

// ---------------------------------------------------------------------------
// degree / norm kernels
// ---------------------------------------------------------------------------
__global__ void init_deg_kernel(float* deg, int n) {
    int i = blockIdx.x * blockDim.x + threadIdx.x;
    if (i < n) deg[i] = 1.0f;   // self-loop contributes 1 to every node's degree
}

__global__ void count_deg_kernel(const int* __restrict__ dst, float* deg, int ne) {
    int e = blockIdx.x * blockDim.x + threadIdx.x;
    if (e < ne) atomAddF32(&deg[dst[e]], 1.0f);
}

__global__ void dinv_kernel(float* dinv, int n) {
    int i = blockIdx.x * blockDim.x + threadIdx.x;
    if (i < n) dinv[i] = rsqrtf(dinv[i]);   // deg >= 1 always (self-loops)
}

__global__ void zero_kernel(float* p, int n) {
    int i = blockIdx.x * blockDim.x + threadIdx.x;
    if (i < n) p[i] = 0.0f;
}

// pack a 4-float group into a v16h fragment (optional scale)
static __device__ __forceinline__ void pack4(v16h& v, int base, float4 f, float s) {
    v[base + 0] = (_Float16)(f.x * s);
    v[base + 1] = (_Float16)(f.y * s);
    v[base + 2] = (_Float16)(f.z * s);
    v[base + 3] = (_Float16)(f.w * s);
}
static __device__ __forceinline__ void pack4n(v16h& v, int base, float4 f) {
    v[base + 0] = (_Float16)f.x;
    v[base + 1] = (_Float16)f.y;
    v[base + 2] = (_Float16)f.z;
    v[base + 3] = (_Float16)f.w;
}

// ---------------------------------------------------------------------------
// WMMA GEMM: out[M,N] = X[M,K] @ W  (+bias, relu optional)
// TRANSB==0: W is [K,N] row-major.  TRANSB==1: W is [N,K] row-major (X @ W^T).
// GUARD==1: N is not a multiple of 16 -> clamp+mask the tail tile columns.
// One wave per 16x16 output tile; f16 WMMA with f32 accumulate.
// Fragment layouts used here (contiguous k order in A and B):
//   A 16x32 f16 : lane 0-15 row=lane (K 0-15), lane 16-31 row=lane-16 (K 16-31)
//   B 32x16 f16 : lane 0-15 col=lane (K 0-15),      lane 16-31 col=lane-16 (K 16-31)
//   C 16x16 f32 : lane gives col (lane&15), vgpr r gives row r+(lane>>4)*8
// K must be a multiple of 32 (true for all layers: 32/64/128).
// ---------------------------------------------------------------------------
template <bool TRANSB, bool GUARD, bool DOBIAS, bool DORELU>
__global__ void wmma_gemm_kernel(const float* __restrict__ X,
                                 const float* __restrict__ W,
                                 const float* __restrict__ bias,
                                 float* __restrict__ out,
                                 int M, int K, int N) {
    const int lane = threadIdx.x & 31;
    const int wave = blockIdx.x * (blockDim.x >> 5) + (threadIdx.x >> 5);
    const int ntn  = (N + 15) >> 4;
    const int tiles = (M >> 4) * ntn;
    if (wave >= tiles) return;
    const int m0 = (wave / ntn) << 4;
    const int n0 = (wave % ntn) << 4;
    const int l15  = lane & 15;
    const int hgrp = lane >> 4;          // 0: lanes 0-15, 1: lanes 16-31

    const int   colB = n0 + l15;
    const int   colC = GUARD ? ((colB < N) ? colB : 0) : colB;
    const float cmsk = GUARD ? ((colB < N) ? 1.0f : 0.0f) : 1.0f;

    const float* xrow = X + (long)(m0 + l15) * K;

    v8f c = {};
    for (int k0 = 0; k0 < K; k0 += 32) {
        // ---- A fragment: one contiguous 16-float run per lane (k = 16*hgrp + e) -> 4x b128 loads
        v16h a;
        {
            const float4* xp = (const float4*)(xrow + k0 + (hgrp << 4));
            float4 x0 = xp[0], x1 = xp[1], x2 = xp[2], x3 = xp[3];
            pack4n(a, 0, x0); pack4n(a, 4, x1);
            pack4n(a, 8, x2); pack4n(a, 12, x3);
        }
        // ---- B fragment
        v16h b;
        const int kb = k0 + (hgrp << 4);
        if (TRANSB) {
            // 16 contiguous floats -> 4x b128 loads
            const float4* wp = (const float4*)(W + (long)colC * K + kb);
            float4 w0 = wp[0], w1 = wp[1], w2 = wp[2], w3 = wp[3];
            if (GUARD) {
                pack4(b, 0, w0, cmsk); pack4(b, 4, w1, cmsk);
                pack4(b, 8, w2, cmsk); pack4(b, 12, w3, cmsk);
            } else {
                pack4n(b, 0, w0); pack4n(b, 4, w1);
                pack4n(b, 8, w2); pack4n(b, 12, w3);
            }
        } else {
            const float* wp = W + (long)kb * N + colC;
            if (GUARD) {
#pragma unroll
                for (int e = 0; e < 16; ++e)
                    b[e] = (_Float16)(wp[(long)e * N] * cmsk);
            } else {
#pragma unroll
                for (int e = 0; e < 16; ++e)
                    b[e] = (_Float16)wp[(long)e * N];
            }
        }
        c = __builtin_amdgcn_wmma_f32_16x16x32_f16(false, a, false, b,
                                                   (short)0, c, false, false);
    }
    // store C (EXEC-masked store for the N<16 tail tile)
    if (!GUARD || colB < N) {
        float badd = DOBIAS ? bias[colB] : 0.0f;
#pragma unroll
        for (int r = 0; r < 8; ++r) {
            int orow = m0 + r + (hgrp << 3);
            float v = c[r] + badd;
            if (DORELU) v = fmaxf(v, 0.0f);
            out[(long)orow * N + colB] = v;
        }
    }
}

// ---------------------------------------------------------------------------
// GCN edge scatter: out[dst] += dinv[src]*dinv[dst] * h[src]   (atomic f32)
// ---------------------------------------------------------------------------
__global__ void scatter_kernel(const int* __restrict__ src,
                               const int* __restrict__ dst,
                               const float* __restrict__ dinv,
                               const float* __restrict__ h,
                               float* __restrict__ out,
                               int ne, int F) {
    int e = blockIdx.x * blockDim.x + threadIdx.x;
    if (e >= ne) return;
    int s = src[e], d = dst[e];
    float w = dinv[s] * dinv[d];
    const float* hs = h + (long)s * F;
    float* od = out + (long)d * F;
    __builtin_prefetch(hs, 0, 0);
#pragma unroll 4
    for (int f = 0; f < F; ++f) atomAddF32(&od[f], w * hs[f]);
}

// finalize: out = relu(agg + dinv^2 * h + bias)
__global__ void finalize_kernel(const float* __restrict__ agg,
                                const float* __restrict__ h,
                                const float* __restrict__ dinv,
                                const float* __restrict__ bias,
                                float* __restrict__ out,
                                int n, int F) {
    int i = blockIdx.x * blockDim.x + threadIdx.x;
    if (i >= n * F) return;
    int node = i / F, f = i - node * F;
    float di = dinv[node];
    float v = agg[i] + di * di * h[i] + bias[f];
    out[i] = fmaxf(v, 0.0f);
}

// ---------------------------------------------------------------------------
// V transpose + f16 convert: vt[d][i] = (f16) qkv[i*96 + 64 + d], d = 0..31.
// One thread per node: 32 contiguous f32 reads, 32 strided f16 writes (once).
// ---------------------------------------------------------------------------
__global__ void vtrans_kernel(const float* __restrict__ qkv,
                              _Float16* __restrict__ vt, int n) {
    int i = blockIdx.x * blockDim.x + threadIdx.x;
    if (i >= n) return;
    const float* vp = qkv + (long)i * 96 + 64;
#pragma unroll
    for (int d = 0; d < 32; ++d)
        vt[(long)d * n + i] = (_Float16)vp[d];
}

// ---------------------------------------------------------------------------
// Flash attention, head_dim=32, 1 head, seq=n.  qkv: [n,96] (q|k|v), out [n,32]
// vt: [32,n] f16 (pre-transposed V).  One wave handles 16 query rows; streams
// 32 keys per iteration.
//
// Transposed formulation: S^T = K.Q^T and out^T = V^T.P put the QUERY index on
// the lane axis of the WMMA C layout, so the softmax running state (m, l,
// alpha) is one scalar per lane, the key-reduction is 16 in-lane VGPRs plus a
// single shfl_xor(16) pair-combine, and the accumulator rescale is lane-local.
// P goes through LDS once per iteration (each lane reads back its own keys) via 2x ds_store_b128 +
// 2x ds_load_b128 directly in B-fragment order.
// ---------------------------------------------------------------------------
__global__ void flash_attn_kernel(const float* __restrict__ qkv,
                                  const _Float16* __restrict__ vt,
                                  float* __restrict__ out, int n) {
    __shared__ __align__(16) _Float16 lds_p[8][16 * 32];   // per-wave P tile
    const int lane = threadIdx.x & 31;
    const int wid  = threadIdx.x >> 5;
    const int wave = blockIdx.x * (blockDim.x >> 5) + wid;
    const int m0 = wave << 4;
    if (m0 >= n) return;
    const int l15  = lane & 15;     // query (col) index of this lane
    const int hgrp = lane >> 4;
    const float scale = 0.17677669529663687f;   // 1/sqrt(32)

    // B fragment: Q^T (kdim x query), loaded once. B[k][q]: k = (e>>3)*16 + hgrp*8 + (e&7)
    v16h bq;
    {
        const float4* qp = (const float4*)(qkv + (long)(m0 + l15) * 96 + (hgrp << 3));
        const float4* qq = (const float4*)(qkv + (long)(m0 + l15) * 96 + 16 + (hgrp << 3));
        float4 x0 = qp[0], x1 = qp[1], x2 = qq[0], x3 = qq[1];
        pack4(bq, 0, x0, scale); pack4(bq, 4, x1, scale);
        pack4(bq, 8, x2, scale); pack4(bq, 12, x3, scale);
    }

    float m_i = -1e30f, l_i = 0.0f;     // per-lane (per-query) softmax state
    v8f acc0 = {}, acc1 = {};           // out^T tiles: d rows 0-15 / 16-31

    _Float16* pw = &lds_p[wid][0];
    const _Float16* vrow0 = vt + (long)l15 * n;        // d = l15      (acc0)
    const _Float16* vrow1 = vt + (long)(16 + l15) * n; // d = 16 + l15 (acc1)

    for (int kb = 0; kb < n; kb += 32) {
        // A fragments: K rows kb+l15 (ak0) and kb+16+l15 (ak1), head-dim 32
        v16h ak0, ak1;
        {
            const float* kr0 = qkv + (long)(kb + l15) * 96 + 32;
            const float* kr1 = qkv + (long)(kb + 16 + l15) * 96 + 32;
            const float4* p0 = (const float4*)(kr0 + (hgrp << 3));
            const float4* p1 = (const float4*)(kr0 + 16 + (hgrp << 3));
            const float4* p2 = (const float4*)(kr1 + (hgrp << 3));
            const float4* p3 = (const float4*)(kr1 + 16 + (hgrp << 3));
            float4 a0 = p0[0], a1 = p0[1], a2 = p1[0], a3 = p1[1];
            float4 b0 = p2[0], b1 = p2[1], b2 = p3[0], b3 = p3[1];
            pack4n(ak0, 0, a0); pack4n(ak0, 4, a1);
            pack4n(ak0, 8, a2); pack4n(ak0, 12, a3);
            pack4n(ak1, 0, b0); pack4n(ak1, 4, b1);
            pack4n(ak1, 8, b2); pack4n(ak1, 12, b3);
        }
        v8f zc = {};
        // S^T tiles: rows = keys (kb + r + 8*hgrp [+16]), cols = queries
        v8f s0 = __builtin_amdgcn_wmma_f32_16x16x32_f16(false, ak0, false, bq,
                                                        (short)0, zc, false, false);
        v8f s1 = __builtin_amdgcn_wmma_f32_16x16x32_f16(false, ak1, false, bq,
                                                        (short)0, zc, false, false);
        // in-lane max over this lane's 16 keys, then combine lane pair (xor 16)
        float mx = s0[0];
#pragma unroll
        for (int r = 1; r < 8; ++r) mx = fmaxf(mx, s0[r]);
#pragma unroll
        for (int r = 0; r < 8; ++r) mx = fmaxf(mx, s1[r]);
        mx = fmaxf(mx, __shfl_xor(mx, 16, 32));
        float mnew = fmaxf(m_i, mx);

        v8h ph0, ph1;
        float sum = 0.0f;
#pragma unroll
        for (int r = 0; r < 8; ++r) {
            float p0 = __expf(s0[r] - mnew);
            float p1 = __expf(s1[r] - mnew);
            sum += p0 + p1;
            ph0[r] = (_Float16)p0;
            ph1[r] = (_Float16)p1;
        }
        sum += __shfl_xor(sum, 16, 32);
        float alpha = __expf(m_i - mnew);
        l_i = l_i * alpha + sum;
        m_i = mnew;
#pragma unroll
        for (int r = 0; r < 8; ++r) { acc0[r] *= alpha; acc1[r] *= alpha; }

        // store P to LDS in B-fragment order: P[k][q] at pw[q*32 + k]
        // this lane holds keys k = 8*hgrp + r (ph0) and 16 + 8*hgrp + r (ph1)
        *(v8h*)(pw + l15 * 32 + (hgrp << 3))      = ph0;
        *(v8h*)(pw + l15 * 32 + 16 + (hgrp << 3)) = ph1;
        __builtin_amdgcn_wave_barrier();
        // B fragment of P: lane (q, hgrp) needs k = (e>>3)*16 + hgrp*8 + (e&7)
        v16h bp;
        {
            const v8h* pr = (const v8h*)(pw + l15 * 32 + (hgrp << 3));
            const v8h* pq = (const v8h*)(pw + l15 * 32 + 16 + (hgrp << 3));
            v8h x = pr[0], y = pq[0];
#pragma unroll
            for (int e = 0; e < 8; ++e) { bp[e] = x[e]; bp[8 + e] = y[e]; }
        }
        __builtin_amdgcn_wave_barrier();
        // A fragments of V^T from pre-transposed f16: contiguous 8-half runs
        // element e -> key k = (e>>3)*16 + hgrp*8 + (e&7)
        v16h av0, av1;
        {
            v8h x0 = *(const v8h*)(vrow0 + kb + (hgrp << 3));
            v8h x1 = *(const v8h*)(vrow0 + kb + 16 + (hgrp << 3));
            v8h y0 = *(const v8h*)(vrow1 + kb + (hgrp << 3));
            v8h y1 = *(const v8h*)(vrow1 + kb + 16 + (hgrp << 3));
#pragma unroll
            for (int e = 0; e < 8; ++e) {
                av0[e] = x0[e]; av0[8 + e] = x1[e];
                av1[e] = y0[e]; av1[8 + e] = y1[e];
            }
        }
        acc0 = __builtin_amdgcn_wmma_f32_16x16x32_f16(false, av0, false, bp,
                                                      (short)0, acc0, false, false);
        acc1 = __builtin_amdgcn_wmma_f32_16x16x32_f16(false, av1, false, bp,
                                                      (short)0, acc1, false, false);
    }
    // out^T(d, q) -> out[(m0+q)*32 + d]; acc0: d = 8*hgrp + r, acc1: +16
    float inv = 1.0f / l_i;
    float* op = out + (long)(m0 + l15) * 32 + (hgrp << 3);
    float4 f0 = { acc0[0] * inv, acc0[1] * inv, acc0[2] * inv, acc0[3] * inv };
    float4 f1 = { acc0[4] * inv, acc0[5] * inv, acc0[6] * inv, acc0[7] * inv };
    float4 f2 = { acc1[0] * inv, acc1[1] * inv, acc1[2] * inv, acc1[3] * inv };
    float4 f3 = { acc1[4] * inv, acc1[5] * inv, acc1[6] * inv, acc1[7] * inv };
    *(float4*)(op)      = f0;
    *(float4*)(op + 4)  = f1;
    *(float4*)(op + 16) = f2;
    *(float4*)(op + 20) = f3;
}

// ---------------------------------------------------------------------------
// host side
// ---------------------------------------------------------------------------
static inline int ceil_div(int a, int b) { return (a + b - 1) / b; }

extern "C" void kernel_launch(void* const* d_in, const int* in_sizes, int n_in,
                              void* d_out, int out_size, void* d_ws, size_t ws_size,
                              hipStream_t stream) {
    const int NN = N_NODES, NE = N_EDGES;
    const int* adj  = (const int*)d_in[0];
    const int* srcE = adj;
    const int* dstE = adj + NE;
    const float* feat = (const float*)d_in[1];
    const float* w1 = (const float*)d_in[2];  const float* b1 = (const float*)d_in[3];
    const float* w2 = (const float*)d_in[4];  const float* b2 = (const float*)d_in[5];
    const float* w3 = (const float*)d_in[6];  const float* b3 = (const float*)d_in[7];
    const float* wqkv = (const float*)d_in[8];  const float* bqkv = (const float*)d_in[9];
    const float* wo   = (const float*)d_in[10]; const float* bo   = (const float*)d_in[11];
    const float* w4 = (const float*)d_in[12]; const float* b4 = (const float*)d_in[13];
    const float* w5a = (const float*)d_in[14]; const float* b5a = (const float*)d_in[15];
    const float* w6a = (const float*)d_in[16]; const float* b6a = (const float*)d_in[17];
    const float* w7a = (const float*)d_in[18]; const float* b7a = (const float*)d_in[19];
    const float* w5f = (const float*)d_in[20]; const float* b5f = (const float*)d_in[21];
    const float* w6f = (const float*)d_in[22]; const float* b6f = (const float*)d_in[23];
    const float* w7f = (const float*)d_in[24]; const float* b7f = (const float*)d_in[25];

    float* ws   = (float*)d_ws;
    float* dinv = ws;                        // 8192
    float* hA   = dinv + NN;                 // 8192*128
    float* hB   = hA + (long)NN * 128;       // 8192*128  (GEMM result)
    float* hC   = hB + (long)NN * 128;       // 8192*128  (aggregation accumulator)
    float* hD   = hC + (long)NN * 128;       // 8192*128  (head intermediates)
    float* qkvb = hD + (long)NN * 128;       // 8192*96
    float* aout = qkvb + (long)NN * 96;      // 8192*32
    _Float16* vt = (_Float16*)(aout + (long)NN * 32);  // 32 x 8192 f16 (V^T)

    auto blocks_for = [&](int M, int N) {
        int tiles = (M >> 4) * ceil_div(N, 16);
        return ceil_div(tiles, 8);           // 8 waves / 256-thread block
    };
    // plain GEMM for GCN layers (no transpose, no bias/relu, N % 16 == 0)
    auto gemm_plain = [&](const float* X, const float* W, float* out, int K, int N) {
        wmma_gemm_kernel<false, false, false, false>
            <<<blocks_for(NN, N), 256, 0, stream>>>(X, W, nullptr, out, NN, K, N);
    };
    // guarded GEMM (N not a multiple of 16; only the N=2 head layer)
    auto gemm_guard = [&](const float* X, const float* W, float* out, int K, int N) {
        wmma_gemm_kernel<false, true, false, false>
            <<<blocks_for(NN, N), 256, 0, stream>>>(X, W, nullptr, out, NN, K, N);
    };
    // X @ W^T + bias (attention projections)
    auto gemm_tb = [&](const float* X, const float* W, const float* bias,
                       float* out, int K, int N) {
        wmma_gemm_kernel<true, false, true, false>
            <<<blocks_for(NN, N), 256, 0, stream>>>(X, W, bias, out, NN, K, N);
    };
    auto gcn = [&](const float* X, const float* W, const float* bias,
                   int K, int Nout, float* target) {
        if (Nout % 16 == 0) gemm_plain(X, W, hB, K, Nout);
        else                gemm_guard(X, W, hB, K, Nout);
        zero_kernel<<<ceil_div(NN * Nout, 256), 256, 0, stream>>>(hC, NN * Nout);
        scatter_kernel<<<ceil_div(NE, 256), 256, 0, stream>>>(srcE, dstE, dinv, hB,
                                                              hC, NE, Nout);
        finalize_kernel<<<ceil_div(NN * Nout, 256), 256, 0, stream>>>(hC, hB, dinv,
                                                                      bias, target,
                                                                      NN, Nout);
    };

    // symmetric normalization with self-loops
    init_deg_kernel<<<ceil_div(NN, 256), 256, 0, stream>>>(dinv, NN);
    count_deg_kernel<<<ceil_div(NE, 256), 256, 0, stream>>>(dstE, dinv, NE);
    dinv_kernel<<<ceil_div(NN, 256), 256, 0, stream>>>(dinv, NN);

    // trunk GCN layers
    gcn(feat, w1, b1, 128, 64, hA);
    gcn(hA,   w2, b2,  64, 32, hA);
    gcn(hA,   w3, b3,  32, 32, hA);

    // self-attention over all nodes
    gemm_tb(hA, wqkv, bqkv, qkvb, 32, 96);
    vtrans_kernel<<<ceil_div(NN, 256), 256, 0, stream>>>(qkvb, vt, NN);
    flash_attn_kernel<<<NN / 16 / 8, 256, 0, stream>>>(qkvb, vt, aout, NN);
    gemm_tb(aout, wo, bo, hA, 32, 32);

    gcn(hA, w4, b4, 32, 32, hA);

    float* da = (float*)d_out;               // [8192, 2]
    float* df = da + (long)NN * 2;           // [8192, 128]

    // head a
    gcn(hA, w5a, b5a, 32, 32, hD);
    gcn(hD, w6a, b6a, 32, 64, hD);
    gcn(hD, w7a, b7a, 64,  2, da);
    // head f
    gcn(hA, w5f, b5f, 32, 32, hD);
    gcn(hD, w6f, b6f, 32, 64, hD);
    gcn(hD, w7f, b7f, 64, 128, df);
}
